// RlSpinn_85753317032665
// MI455X (gfx1250) — matrix operand order repaired and checked by reading, not compiled, not run on hardware
//
#include <hip/hip_runtime.h>
#include <hip/hip_bf16.h>
#include <math.h>

// Problem constants (from reference): B=512, L=32, D=DT=128, S=2L-1=63
#define BB   512
#define LL   32
#define DD   128
#define DTT  128
#define SSL  63
#define BT   16              // batch rows per workgroup (== WMMA M)
#define NEGF (-3.4028234663852886e38f)

typedef __attribute__((ext_vector_type(16))) __bf16 v16bf;
typedef __attribute__((ext_vector_type(8)))  __bf16 v8bf;
typedef __attribute__((ext_vector_type(8)))  float  v8f;

__device__ __forceinline__ int imod(int a, int m) { int r = a % m; return r < 0 ? r + m : r; }
__device__ __forceinline__ float sigf(float x) { return 1.0f / (1.0f + expf(-x)); }

// ---------------- shared memory layout (~58 KB) ----------------
struct __align__(32) SMem {
  __bf16 X[BT * 512];        // A-matrix staging (track: [16][512]; comp reuses as [16][256])
  float  h[BT * DTT];        // tracker h
  float  c[BT * DTT];        // tracker c
  float  S1C[BT * DD];       // st1 c-half; later aliased as NEW[:, :128] (=ch)
  float  S2C[BT * DD];       // st2 c-half; later aliased as NEW[:, 128:] (=cc)
  float  track_b[4 * DTT];
  float  comp_b[5 * DD];
  float  TW[DTT * 2];        // trans_W row-major [128][2]
  float  tb[2];
  float  lng[2 * DD];
  float  lnb[2 * DD];
  float  logit[BT][2];
  float  mu[BT], rs[BT];
  int    queues[BT][LL + 1];
  int    qi[BT], bp[BT], us[BT], len[BT], qtop[BT], rightq[BT], isr[BT];
};

__device__ __forceinline__ float newval(SMem& sm, int r, int col) {
  return (col < DD) ? sm.S1C[r * DD + col] : sm.S2C[r * DD + (col - DD)];
}

__device__ __forceinline__ void ln_stats(SMem& sm, int tid) {
  if (tid < BT) {
    int r = tid;
    float s = 0.0f;
    for (int cI = 0; cI < 2 * DD; ++cI) s += newval(sm, r, cI);
    float mu = s * (1.0f / (2 * DD));
    float v = 0.0f;
    for (int cI = 0; cI < 2 * DD; ++cI) { float d = newval(sm, r, cI) - mu; v += d * d; }
    v *= (1.0f / (2 * DD));
    sm.mu[r] = mu;
    sm.rs[r] = rsqrtf(v + 1e-5f);
  }
}

// ---------------- weight transpose + cast to bf16 ----------------
__global__ void cast_trackW(const float* __restrict__ W, __bf16* __restrict__ WT) {
  int i = blockIdx.x * blockDim.x + threadIdx.x;          // over 512*512
  if (i >= 512 * 512) return;
  int n = i >> 9, k = i & 511;                            // WT[n][k] = W[k][n]
  WT[i] = (__bf16)W[k * 512 + n];
}
__global__ void cast_compW(const float* __restrict__ W, __bf16* __restrict__ WT) {
  int i = blockIdx.x * blockDim.x + threadIdx.x;          // over 640*256
  if (i >= 640 * 256) return;
  int n = i >> 8, k = i & 255;                            // WT[n][k] = W[k][n]
  WT[i] = (__bf16)W[k * 640 + n];
}

// ---------------- persistent SPINN kernel: one WG = 16 batch rows, full scan ----------------
__global__ void __launch_bounds__(128)
spinn_kernel(const int*   __restrict__ messages,
             const float* __restrict__ embedding,
             const __bf16* __restrict__ trackWT,   // [512 N][512 K] bf16
             const __bf16* __restrict__ compWT,    // [640 N][256 K] bf16
             const float* __restrict__ track_b,
             const float* __restrict__ trans_W,
             const float* __restrict__ trans_b,
             const float* __restrict__ comp_b,
             const float* __restrict__ ln_g,
             const float* __restrict__ ln_b,
             float* __restrict__ out)
{
  __shared__ SMem sm;
  const int tid  = threadIdx.x;
  const int row0 = blockIdx.x * BT;

  float* outHidden = out;
  float* thinG     = out + (size_t)BB * DD;
  float* outTrans  = thinG + (size_t)BB * SSL * 2 * DD;
  float* outLogp   = outTrans + (size_t)BB * SSL;
  float* outEnt    = outLogp  + (size_t)BB * SSL;

  auto thinRow = [&](int r, int s) -> float* {
    return thinG + ((size_t)(row0 + r) * SSL + s) * (2 * DD);
  };

  // ---- stage small weights/biases into LDS
  for (int i = tid; i < 4 * DTT; i += 128) sm.track_b[i] = track_b[i];
  for (int i = tid; i < 5 * DD;  i += 128) sm.comp_b[i]  = comp_b[i];
  for (int i = tid; i < DTT * 2; i += 128) sm.TW[i]      = trans_W[i];
  for (int i = tid; i < 2 * DD;  i += 128) { sm.lng[i] = ln_g[i]; sm.lnb[i] = ln_b[i]; }
  if (tid < 2) sm.tb[tid] = trans_b[tid];

  // ---- per-row state init
  if (tid < BT) {
    int r = tid, g = row0 + r;
    int len = LL;
    for (int i = 0; i < LL; ++i)
      if (messages[g * LL + i] == 0) { len = i + 1; break; }
    sm.len[r] = len;
    sm.qi[r] = 1; sm.bp[r] = 2; sm.us[r] = len - 2;
    for (int i = 0; i < LL + 1; ++i) sm.queues[r][i] = -1;
    sm.queues[r][0] = 0; sm.queues[r][1] = 1;
  }
  for (int i = tid; i < BT * DTT; i += 128) { sm.h[i] = 0.0f; sm.c[i] = 0.0f; }

  // ---- zero this WG's thin region (reference thin starts as zeros)
  {
    float* base = thinG + (size_t)row0 * SSL * 2 * DD;
    for (int i = tid; i < BT * SSL * 2 * DD; i += 128) base[i] = 0.0f;
  }
  __syncthreads();

  // ---- thin[:,0] / thin[:,1] = LN(reps[:,0/1])
  for (int slot = 0; slot < 2; ++slot) {
    for (int i = tid; i < BT * 2 * DD; i += 128) {
      int r = i >> 8, col = i & 255;
      int idx = messages[(row0 + r) * LL + slot];
      float v = embedding[(size_t)idx * (2 * DD) + col];
      if (col < DD) sm.S1C[r * DD + col] = v; else sm.S2C[r * DD + (col - DD)] = v;
    }
    __syncthreads();
    ln_stats(sm, tid);
    __syncthreads();
    for (int i = tid; i < BT * 2 * DD; i += 128) {
      int r = i >> 8, col = i & 255;
      float x = newval(sm, r, col);
      thinRow(r, slot)[col] = (x - sm.mu[r]) * sm.rs[r] * sm.lng[col] + sm.lnb[col];
    }
    __threadfence();
    __syncthreads();
  }

  // ---- output columns 0,1 (SHIFT, masked by done)
  if (tid < BT) {
    int r = tid, g = row0 + r;
    for (int col = 0; col < 2; ++col) {
      bool done = col >= (2 * sm.len[r] - 1);
      outTrans[(size_t)g * SSL + col] = done ? 2.0f : 0.0f;
      outLogp[(size_t)g * SSL + col]  = 0.0f;
      outEnt[(size_t)g * SSL + col]   = 0.0f;
    }
  }

  const int wv = tid >> 5, lane = tid & 31, m = lane & 15, hi = lane >> 4;

  // ================= main scan: t = 2 .. 62 =================
  for (int t = 2; t < SSL; ++t) {
    if (tid < BT) {
      int r = tid;
      sm.qtop[r] = sm.queues[r][imod(sm.qi[r], LL + 1)];
    }
    __syncthreads();

    // ---- build A for tracker: X = [top_buf | s1 | s2 | h] as bf16 [16][512]
    for (int i = tid; i < BT * 512; i += 128) {
      int r = i >> 9, col = i & 511;
      float v;
      if (col < DD) {
        int idx = messages[(row0 + r) * LL + (sm.bp[r] % LL)];
        v = embedding[(size_t)idx * (2 * DD) + col];
      } else if (col < 2 * DD) {
        v = thinRow(r, imod(sm.qtop[r], SSL))[col - DD];
      } else if (col < 3 * DD) {
        v = thinRow(r, imod(sm.qtop[r] - 1, SSL))[col - 2 * DD];
      } else {
        v = sm.h[r * DTT + (col - 3 * DD)];
      }
      sm.X[i] = (__bf16)v;
    }
    __syncthreads();

    // ---- tracker GEMM [16,512]@[512,512] via v_wmma_f32_16x16x32_bf16 + in-register LSTM gates
    {
      v8f acc[8] = {};
      for (int ks = 0; ks < 16; ++ks) {
        union { v16bf v; v8bf p[2]; } a;
        a.p[0] = *(const v8bf*)&sm.X[m * 512 + ks * 32 + 16 * hi];
        a.p[1] = *(const v8bf*)&sm.X[m * 512 + ks * 32 + 8 + 16 * hi];
        #pragma unroll
        for (int j = 0; j < 8; ++j) {
          int grp = 2 * wv + (j >> 2);     // gate-column group 0..7
          int gb  = j & 3;                 // gate block i/f/o/g
          int nt  = gb * 8 + grp;          // N tile 0..31
          v16bf bv = *(const v16bf*)&trackWT[(size_t)(nt * 16 + m) * 512 + ks * 32 + 16 * hi];
          acc[j] = __builtin_amdgcn_wmma_f32_16x16x32_bf16(
              false, a.v, false, bv, (short)0, acc[j], false, false);
        }
      }
      #pragma unroll
      for (int g2 = 0; g2 < 2; ++g2) {
        int col = 16 * (2 * wv + g2) + m;        // d in [0,128)
        #pragma unroll
        for (int r8 = 0; r8 < 8; ++r8) {
          int M = r8 + 8 * hi;
          float gi_ = acc[g2 * 4 + 0][r8] + sm.track_b[col];
          float gf_ = acc[g2 * 4 + 1][r8] + sm.track_b[DTT + col];
          float go_ = acc[g2 * 4 + 2][r8] + sm.track_b[2 * DTT + col];
          float gg_ = acc[g2 * 4 + 3][r8] + sm.track_b[3 * DTT + col];
          float cn = sigf(gf_) * sm.c[M * DTT + col] + sigf(gi_) * tanhf(gg_);
          float hn = sigf(go_) * tanhf(cn);
          sm.c[M * DTT + col] = cn;
          sm.h[M * DTT + col] = hn;
        }
      }
    }
    __syncthreads();

    // ---- transition logits (tiny 128x2)
    if (tid < 2 * BT) {
      int r = tid >> 1, j = tid & 1;
      float s = sm.tb[j];
      for (int k = 0; k < DTT; ++k) s += sm.h[r * DTT + k] * sm.TW[k * 2 + j];
      sm.logit[r][j] = s;
    }
    __syncthreads();

    // ---- per-row decision + stack bookkeeping
    if (tid < BT) {
      int r = tid, g = row0 + r;
      float l0 = sm.logit[r][0] + ((sm.us[r] == 0) ? NEGF : 0.0f);
      float l1 = sm.logit[r][1] + ((sm.qi[r] <= 0) ? NEGF : 0.0f);
      float mx = fmaxf(l0, l1);
      float lz = logf(expf(l0 - mx) + expf(l1 - mx));
      float lp0 = l0 - mx - lz, lp1 = l1 - mx - lz;
      int trans = (l1 > l0) ? 1 : 0;
      float log_prob = trans ? lp1 : lp0;
      float p0 = expf(lp0), p1 = expf(lp1);
      float ent = 0.0f;
      if (p0 > 0.0f) ent -= p0 * lp0;
      if (p1 > 0.0f) ent -= p1 * lp1;
      int is_s = (trans == 0) ? 1 : 0;
      int is_r = (trans == 1) ? 1 : 0;
      int qi = sm.qi[r] + is_s - is_r;
      qi = qi < -1 ? -1 : (qi > LL - 1 ? LL - 1 : qi);
      int bp = sm.bp[r] + is_s; if (bp > LL - 1) bp = LL - 1;
      sm.queues[r][imod(qi, LL + 1)] = t;
      int nxt = imod(qi + 1, LL + 1);
      if (is_r) sm.queues[r][nxt] = -1;
      sm.qi[r] = qi; sm.bp[r] = bp; sm.us[r] -= is_s;
      sm.rightq[r] = sm.queues[r][nxt];
      sm.isr[r] = is_r;
      bool done = t >= (2 * sm.len[r] - 1);
      outTrans[(size_t)g * SSL + t] = done ? 2.0f : (float)trans;
      outLogp[(size_t)g * SSL + t]  = done ? 0.0f : log_prob;
      outEnt[(size_t)g * SSL + t]   = done ? 0.0f : ent;
    }
    __syncthreads();

    // ---- build comp A: X2 = [s1h+s2h | h] bf16 [16][256]; stash c-halves
    for (int i = tid; i < BT * 2 * DD; i += 128) {
      int r = i >> 8, col = i & 255;
      if (col < DD) {
        const float* st1 = thinRow(r, imod(t, SSL));                 // left == t (stale row)
        const float* st2 = thinRow(r, imod(sm.rightq[r], SSL));
        sm.X[r * 256 + col]  = (__bf16)(st1[col] + st2[col]);
        sm.S1C[r * DD + col] = st1[DD + col];
        sm.S2C[r * DD + col] = st2[DD + col];
      } else {
        sm.X[r * 256 + col] = (__bf16)sm.h[r * DTT + (col - DD)];
      }
    }
    __syncthreads();

    // ---- comp GEMM [16,256]@[256,640] + in-register TreeLSTM; NEW overwrites S1C/S2C
    {
      v8f acc[10] = {};
      for (int ks = 0; ks < 8; ++ks) {
        union { v16bf v; v8bf p[2]; } a;
        a.p[0] = *(const v8bf*)&sm.X[m * 256 + ks * 32 + 16 * hi];
        a.p[1] = *(const v8bf*)&sm.X[m * 256 + ks * 32 + 8 + 16 * hi];
        #pragma unroll
        for (int j = 0; j < 10; ++j) {
          int grp = 2 * wv + (j >= 5 ? 1 : 0);
          int gb  = j % 5;                 // ci/cfl/cfr/co/cg
          int nt  = gb * 8 + grp;          // N tile 0..39
          v16bf bv = *(const v16bf*)&compWT[(size_t)(nt * 16 + m) * 256 + ks * 32 + 16 * hi];
          acc[j] = __builtin_amdgcn_wmma_f32_16x16x32_bf16(
              false, a.v, false, bv, (short)0, acc[j], false, false);
        }
      }
      #pragma unroll
      for (int g2 = 0; g2 < 2; ++g2) {
        int col = 16 * (2 * wv + g2) + m;
        #pragma unroll
        for (int r8 = 0; r8 < 8; ++r8) {
          int M = r8 + 8 * hi;
          float ci_  = acc[g2 * 5 + 0][r8] + sm.comp_b[col];
          float cfl_ = acc[g2 * 5 + 1][r8] + sm.comp_b[DD + col];
          float cfr_ = acc[g2 * 5 + 2][r8] + sm.comp_b[2 * DD + col];
          float co_  = acc[g2 * 5 + 3][r8] + sm.comp_b[3 * DD + col];
          float cg_  = acc[g2 * 5 + 4][r8] + sm.comp_b[4 * DD + col];
          float s1c = sm.S1C[M * DD + col], s2c = sm.S2C[M * DD + col];
          float cc = sigf(cfl_) * s2c + sigf(cfr_) * s1c + sigf(ci_) * tanhf(cg_);
          float ch = sigf(co_) * cc;
          float n0, n1;
          if (sm.isr[M]) { n0 = ch; n1 = cc; }
          else {
            int idx = messages[(row0 + M) * LL + (sm.bp[M] % LL)];
            n0 = embedding[(size_t)idx * (2 * DD) + col];
            n1 = embedding[(size_t)idx * (2 * DD) + DD + col];
          }
          sm.S1C[M * DD + col] = n0;    // NEW[:, :128]
          sm.S2C[M * DD + col] = n1;    // NEW[:, 128:]
        }
      }
    }
    __syncthreads();

    // ---- LayerNorm(NEW) -> thin[:, t]
    ln_stats(sm, tid);
    __syncthreads();
    for (int i = tid; i < BT * 2 * DD; i += 128) {
      int r = i >> 8, col = i & 255;
      float x = newval(sm, r, col);
      thinRow(r, t)[col] = (x - sm.mu[r]) * sm.rs[r] * sm.lng[col] + sm.lnb[col];
    }
    __threadfence();
    __syncthreads();
  }

  // ---- hidden = thin[row, 2*len-2][:128]
  for (int i = tid; i < BT * DD; i += 128) {
    int r = i >> 7, d = i & 127;
    outHidden[(size_t)(row0 + r) * DD + d] = thinRow(r, 2 * sm.len[r] - 2)[d];
  }
}

extern "C" void kernel_launch(void* const* d_in, const int* in_sizes, int n_in,
                              void* d_out, int out_size, void* d_ws, size_t ws_size,
                              hipStream_t stream) {
  (void)in_sizes; (void)n_in; (void)out_size; (void)ws_size;
  const int*   messages  = (const int*)d_in[0];
  const float* embedding = (const float*)d_in[1];
  const float* track_W   = (const float*)d_in[2];
  const float* track_b   = (const float*)d_in[3];
  const float* trans_W   = (const float*)d_in[4];
  const float* trans_b   = (const float*)d_in[5];
  const float* comp_W    = (const float*)d_in[6];
  const float* comp_b    = (const float*)d_in[7];
  const float* ln_g      = (const float*)d_in[8];
  const float* ln_b      = (const float*)d_in[9];
  float* out = (float*)d_out;

  __bf16* trackWT = (__bf16*)d_ws;                 // 512*512 bf16 = 512 KB
  __bf16* compWT  = trackWT + 512 * 512;           // 640*256 bf16 = 320 KB

  cast_trackW<<<(512 * 512 + 255) / 256, 256, 0, stream>>>(track_W, trackWT);
  cast_compW <<<(640 * 256 + 255) / 256, 256, 0, stream>>>(comp_W, compWT);
  spinn_kernel<<<BB / BT, 128, 0, stream>>>(messages, embedding, trackWT, compWT,
                                            track_b, trans_W, trans_b, comp_b,
                                            ln_g, ln_b, out);
}
